// MoELayer_17411797418663
// MI455X (gfx1250) — compile-verified, matrix operand order checked by reading, not run on hardware
//
#include <hip/hip_runtime.h>
#include <cstdint>

// MoE top-2: T=2048 tokens, D=512, E=16 experts, H=2048, fp32 in/out.
// route -> gather per expert -> bf16 WMMA GEMM1 (+b1, SiLU) -> LayerNorm ->
// bf16 WMMA GEMM2 (+b2, *routing weight, atomic scatter-add).
// GEMMs: 128x128 tiles, 8 waves/block, wave = 64x32 (4x2 WMMA 16x16x32 bf16),
// double-buffered LDS, software-pipelined global->reg->LDS staging.
// Workspace use: ~48.2 MB.

#define T_TOK 2048
#define DIMD  512
#define NE    16
#define HID   2048
#define LN_EPS 1e-5f

typedef __attribute__((ext_vector_type(16))) __bf16   v16bf;
typedef __attribute__((ext_vector_type(8)))  float    v8f;
typedef __attribute__((ext_vector_type(4)))  float    f32x4;
typedef __attribute__((ext_vector_type(2)))  uint32_t u32x2;
typedef __attribute__((ext_vector_type(4)))  uint32_t u32x4;

union BF16Frag { u32x4 u[2]; v16bf v; };
union F4 { f32x4 v; float a[4]; };

// RNE fp32->bf16 pack done entirely in 32-bit ALU (avoids 16-bit half-reg ops)
__device__ __forceinline__ uint32_t pack2bf(float a, float b) {
  union { float f; uint32_t u; } va, vb;
  va.f = a; vb.f = b;
  uint32_t ra = va.u + 0x7FFFu + ((va.u >> 16) & 1u);
  uint32_t rb = vb.u + 0x7FFFu + ((vb.u >> 16) & 1u);
  return (ra >> 16) | (rb & 0xFFFF0000u);
}

// Force global (addrspace 1) vector loads/stores for staging traffic so the
// backend emits global_load_b128 (LOADcnt only) instead of flat_load_b128
// (which also ties up the LDS/DScnt path). POD ext_vector types only.
typedef const __attribute__((address_space(1))) f32x4* gcf4p;
typedef const __attribute__((address_space(1))) u32x4* gcu4p;
typedef       __attribute__((address_space(1))) u32x4* gu4p;

__device__ __forceinline__ f32x4 gload_f4(const float* p)  { return *(gcf4p)p; }
__device__ __forceinline__ u32x4 gload_u4(const __bf16* p) { return *(gcu4p)p; }
__device__ __forceinline__ void  gstore_u4(__bf16* p, u32x4 v) { *(gu4p)p = v; }

// ---------------- init: zero output + routing counters ----------------
__global__ __launch_bounds__(256) void moe_init_kernel(float* out, int* counts, int* cursor) {
  int i = blockIdx.x * 256 + threadIdx.x;
  if (i < T_TOK * DIMD) out[i] = 0.0f;
  if (i < NE) { counts[i] = 0; cursor[i] = 0; }
}

// ---------------- routing: gate logits, top-2, softmax ----------------
__global__ __launch_bounds__(256) void moe_route_kernel(
    const float* __restrict__ x, const float* __restrict__ Wg,
    int* __restrict__ sel, float* __restrict__ wts, int* __restrict__ counts) {
  int tid = threadIdx.x;
  int tl = tid >> 4, e = tid & 15;
  int t = blockIdx.x * 16 + tl;
  const float* xr = x + (size_t)t * DIMD;
  float acc = 0.0f;
  for (int d = 0; d < DIMD; ++d) acc += xr[d] * Wg[d * NE + e];
  __shared__ float lg[256];
  lg[tid] = acc;
  __syncthreads();
  if (e == 0) {
    float b0 = -3.4e38f, b1v = -3.4e38f; int i0 = 0, i1 = 0;
    for (int j = 0; j < NE; ++j) {
      float v = lg[tl * 16 + j];
      if (v > b0)       { b1v = b0; i1 = i0; b0 = v; i0 = j; }
      else if (v > b1v) { b1v = v; i1 = j; }
    }
    float p0 = 1.0f / (1.0f + __expf(b1v - b0));  // 2-way softmax
    sel[t * 2 + 0] = i0; sel[t * 2 + 1] = i1;
    wts[t * 2 + 0] = p0; wts[t * 2 + 1] = 1.0f - p0;
    atomicAdd(&counts[i0], 1);
    atomicAdd(&counts[i1], 1);
  }
}

// ---------------- prefix over 16 expert counts ----------------
__global__ void moe_prefix_kernel(const int* __restrict__ counts, int* __restrict__ seg) {
  if (threadIdx.x == 0) {
    int s = 0;
    for (int e = 0; e < NE; ++e) { seg[e] = s; s += counts[e]; }
  }
}

// ---------------- scatter pairs into per-expert segments ----------------
__global__ __launch_bounds__(256) void moe_scatter_kernel(
    const int* __restrict__ sel, const float* __restrict__ wts,
    const int* __restrict__ seg, int* __restrict__ cursor,
    int* __restrict__ ptok, float* __restrict__ pw, int* __restrict__ pexp) {
  int p = blockIdx.x * 256 + threadIdx.x;     // 0..4095
  int t = p >> 1, k = p & 1;
  int e = sel[t * 2 + k];
  int pos = seg[e] + atomicAdd(&cursor[e], 1);
  ptok[pos] = t;
  pw[pos]   = wts[t * 2 + k];
  pexp[pos] = e;
}

// LDS row pitch for bf16 tiles (48 elems = 96B, 16B aligned)
#define AKP 48

// A/B fragment loader per ISA 7.12.2 (16-bit, 16x32):
// lanes 0-15: K 0-7 in v0-3, K 16-23 in v4-7; lanes 16-31: K 8-15 / 24-31.
__device__ __forceinline__ v16bf load_frag16(const __bf16* base, int row, int kOff) {
  const u32x4* p = (const u32x4*)(base + row * AKP + kOff);
  BF16Frag f;
  f.u[0] = p[0];   // 8 bf16 at kOff
  f.u[1] = p[2];   // 8 bf16 at kOff+16
  return f.v;
}

// ---------------- GEMM1: h = silu(gather(x) @ W1[e] + b1[e]) ----------------
__global__ __launch_bounds__(256) void moe_gemm1_kernel(
    const float* __restrict__ x, const float* __restrict__ W1,
    const float* __restrict__ b1,
    const int* __restrict__ counts, const int* __restrict__ seg,
    const int* __restrict__ ptok, float* __restrict__ h) {
  int e = blockIdx.z;
  int cnt = counts[e];
  int m0 = blockIdx.y * 128;
  if (m0 >= cnt) return;
  int n0 = blockIdx.x * 128;
  int segs = seg[e];
  int tid = threadIdx.x;

  __shared__ __bf16 As[2][128 * AKP];
  __shared__ __bf16 Bs[2][128 * AKP];
  __shared__ int stok[128];

  if (tid < 128) {
    int r = m0 + tid;
    stok[tid] = (r < cnt) ? ptok[segs + r] : 0;
  }
  __syncthreads();

  const float* W1e = W1 + (size_t)e * DIMD * HID;

  // ---- per-thread staging plan (loop-invariant addressing) ----
  // A tile: 128 rows x 32 K fp32 -> 1024 float4 items, 4 per thread
  int aoff[4];                 // LDS dest (element index)
  const float* aptr[4];        // global src, advance +32 per k-step
#pragma unroll
  for (int i = 0; i < 4; ++i) {
    int fid = tid + i * 256;
    int row = fid >> 3, c4 = fid & 7;
    aoff[i] = row * AKP + c4 * 4;
    aptr[i] = x + (size_t)stok[row] * DIMD + c4 * 4;
  }
  // B tile transposed: Bs[n][k]; 16 k-pairs x 32 n4-groups = 512 items, 2/thread
  int boff[2][4];              // LDS dest per n of the group
  const float* bptr[2];        // global src at W[k][n], advance +32*HID
#pragma unroll
  for (int i = 0; i < 2; ++i) {
    int fid = tid + i * 256;   // 0..511
    int kp = fid >> 5, g = fid & 31;
    int n = g * 4;
#pragma unroll
    for (int j = 0; j < 4; ++j) boff[i][j] = (n + j) * AKP + kp * 2;
    bptr[i] = W1e + (size_t)(kp * 2) * HID + n0 + n;
  }

  int lane = tid & 31, wave = tid >> 5;
  int waveM = (wave >> 2) * 64;
  int waveN = (wave & 3) * 32;
  int laneHalf = lane >> 4, laneM = lane & 15;
  int kOff = laneHalf ? 8 : 0;

  v8f zero = {0, 0, 0, 0, 0, 0, 0, 0};
  v8f acc[4][2];
#pragma unroll
  for (int i = 0; i < 4; ++i)
#pragma unroll
    for (int j = 0; j < 2; ++j) acc[i][j] = zero;

  F4 areg[4], breg[2][2];
  const int NK = DIMD / 32;

  // prologue: load + store k-step 0
#pragma unroll
  for (int i = 0; i < 4; ++i) { areg[i].v = gload_f4(aptr[i]); aptr[i] += 32; }
#pragma unroll
  for (int i = 0; i < 2; ++i) {
    breg[i][0].v = gload_f4(bptr[i]);
    breg[i][1].v = gload_f4(bptr[i] + HID);
    bptr[i] += 32 * HID;
  }
#pragma unroll
  for (int i = 0; i < 4; ++i) {
    u32x2 st = {pack2bf(areg[i].a[0], areg[i].a[1]),
                pack2bf(areg[i].a[2], areg[i].a[3])};
    *(u32x2*)&As[0][aoff[i]] = st;
  }
#pragma unroll
  for (int i = 0; i < 2; ++i)
#pragma unroll
    for (int j = 0; j < 4; ++j)
      *(uint32_t*)&Bs[0][boff[i][j]] = pack2bf(breg[i][0].a[j], breg[i][1].a[j]);
  __syncthreads();

  for (int kt = 0; kt < NK; ++kt) {
    int cur = kt & 1;
    bool hasNext = (kt + 1) < NK;
    if (hasNext) {  // issue next global loads early (overlap with WMMA)
#pragma unroll
      for (int i = 0; i < 4; ++i) { areg[i].v = gload_f4(aptr[i]); aptr[i] += 32; }
#pragma unroll
      for (int i = 0; i < 2; ++i) {
        breg[i][0].v = gload_f4(bptr[i]);
        breg[i][1].v = gload_f4(bptr[i] + HID);
        bptr[i] += 32 * HID;
      }
    }
    v16bf fb[2];
#pragma unroll
    for (int ns = 0; ns < 2; ++ns)
      fb[ns] = load_frag16(Bs[cur], waveN + ns * 16 + laneM, kOff);
#pragma unroll
    for (int ms = 0; ms < 4; ++ms) {
      v16bf fa = load_frag16(As[cur], waveM + ms * 16 + laneM, kOff);
#pragma unroll
      for (int ns = 0; ns < 2; ++ns)
        acc[ms][ns] = __builtin_amdgcn_wmma_f32_16x16x32_bf16(
            false, fa, false, fb[ns], (short)0, acc[ms][ns], false, false);
    }
    if (hasNext) {
      int nxt = cur ^ 1;
#pragma unroll
      for (int i = 0; i < 4; ++i) {
        u32x2 st = {pack2bf(areg[i].a[0], areg[i].a[1]),
                    pack2bf(areg[i].a[2], areg[i].a[3])};
        *(u32x2*)&As[nxt][aoff[i]] = st;
      }
#pragma unroll
      for (int i = 0; i < 2; ++i)
#pragma unroll
        for (int j = 0; j < 4; ++j)
          *(uint32_t*)&Bs[nxt][boff[i][j]] = pack2bf(breg[i][0].a[j], breg[i][1].a[j]);
    }
    __syncthreads();
  }

  // epilogue: + b1, SiLU, store fp32 h
#pragma unroll
  for (int ms = 0; ms < 4; ++ms) {
#pragma unroll
    for (int ns = 0; ns < 2; ++ns) {
      int nG = n0 + waveN + ns * 16 + laneM;
      float bb = b1[e * HID + nG];
      int rBase = waveM + ms * 16 + laneHalf * 8;
#pragma unroll
      for (int r = 0; r < 8; ++r) {
        int row = rBase + r;
        if (m0 + row < cnt) {
          float c = acc[ms][ns][r] + bb;
          float s = c / (1.0f + __expf(-c));
          h[(size_t)(segs + m0 + row) * HID + nG] = s;
        }
      }
    }
  }
}

// ---------------- LayerNorm over H per gathered row, emit bf16 ----------------
// Each thread owns 8 contiguous columns -> one packed b128 bf16 store.
__global__ __launch_bounds__(256) void moe_ln_kernel(
    const float* __restrict__ h, const float* __restrict__ ln_g,
    const float* __restrict__ ln_b, const int* __restrict__ pexp,
    __bf16* __restrict__ hb) {
  int row = blockIdx.x;
  int tid = threadIdx.x;
  int e = pexp[row];
  int c0 = tid * 8;
  const float* hr = h + (size_t)row * HID + c0;
  F4 v0, v1;
  v0.v = gload_f4(hr);
  v1.v = gload_f4(hr + 4);
  float s = 0.0f, s2 = 0.0f;
#pragma unroll
  for (int j = 0; j < 4; ++j) { s += v0.a[j] + v1.a[j];
                                s2 += v0.a[j] * v0.a[j] + v1.a[j] * v1.a[j]; }
  __shared__ float rs[256], rq[256];
  rs[tid] = s; rq[tid] = s2;
  __syncthreads();
  for (int off = 128; off > 0; off >>= 1) {
    if (tid < off) { rs[tid] += rs[tid + off]; rq[tid] += rq[tid + off]; }
    __syncthreads();
  }
  float mu  = rs[0] * (1.0f / HID);
  float var = rq[0] * (1.0f / HID) - mu * mu;
  float inv = rsqrtf(var + LN_EPS);
  F4 g0, g1, be0, be1;
  g0.v  = gload_f4(ln_g + e * HID + c0);
  g1.v  = gload_f4(ln_g + e * HID + c0 + 4);
  be0.v = gload_f4(ln_b + e * HID + c0);
  be1.v = gload_f4(ln_b + e * HID + c0 + 4);
  float o[8];
#pragma unroll
  for (int j = 0; j < 4; ++j) {
    o[j]     = (v0.a[j] - mu) * inv * g0.a[j] + be0.a[j];
    o[j + 4] = (v1.a[j] - mu) * inv * g1.a[j] + be1.a[j];
  }
  u32x4 st = {pack2bf(o[0], o[1]), pack2bf(o[2], o[3]),
              pack2bf(o[4], o[5]), pack2bf(o[6], o[7])};
  gstore_u4(hb + (size_t)row * HID + c0, st);
}

// ---------------- GEMM2: out += w * (h @ W2[e] + b2[e]) ----------------
__global__ __launch_bounds__(256) void moe_gemm2_kernel(
    const __bf16* __restrict__ hb, const float* __restrict__ W2,
    const float* __restrict__ b2,
    const int* __restrict__ counts, const int* __restrict__ seg,
    const int* __restrict__ ptok, const float* __restrict__ pw,
    float* __restrict__ out) {
  int e = blockIdx.z;
  int cnt = counts[e];
  int m0 = blockIdx.y * 128;
  if (m0 >= cnt) return;
  int n0 = blockIdx.x * 128;
  int segs = seg[e];
  int tid = threadIdx.x;

  __shared__ __bf16 As[2][128 * AKP];
  __shared__ __bf16 Bs[2][128 * AKP];
  __shared__ int   stok[128];
  __shared__ float sw[128];

  if (tid < 128) {
    int r = m0 + tid;
    bool ok = r < cnt;
    stok[tid] = ok ? ptok[segs + r] : 0;
    sw[tid]   = ok ? pw[segs + r]   : 0.0f;
  }
  __syncthreads();

  const float* W2e = W2 + (size_t)e * HID * DIMD;

  // A tile: 128 rows x 32 K bf16 = 512 uint4 items, 2 per thread (direct copy)
  int aoff[2];
  const __bf16* aptr[2];
#pragma unroll
  for (int i = 0; i < 2; ++i) {
    int fid = tid + i * 256;   // 0..511
    int row = fid >> 2, c = fid & 3;
    int rr = m0 + row;
    int p = (rr < cnt) ? (segs + rr) : segs;
    aoff[i] = row * AKP + c * 8;
    aptr[i] = hb + (size_t)p * HID + c * 8;
  }
  // B tile transposed: same plan as GEMM1, row stride DIMD
  int boff[2][4];
  const float* bptr[2];
#pragma unroll
  for (int i = 0; i < 2; ++i) {
    int fid = tid + i * 256;
    int kp = fid >> 5, g = fid & 31;
    int n = g * 4;
#pragma unroll
    for (int j = 0; j < 4; ++j) boff[i][j] = (n + j) * AKP + kp * 2;
    bptr[i] = W2e + (size_t)(kp * 2) * DIMD + n0 + n;
  }

  int lane = tid & 31, wave = tid >> 5;
  int waveM = (wave >> 2) * 64;
  int waveN = (wave & 3) * 32;
  int laneHalf = lane >> 4, laneM = lane & 15;
  int kOff = laneHalf ? 8 : 0;

  v8f zero = {0, 0, 0, 0, 0, 0, 0, 0};
  v8f acc[4][2];
#pragma unroll
  for (int i = 0; i < 4; ++i)
#pragma unroll
    for (int j = 0; j < 2; ++j) acc[i][j] = zero;

  u32x4 areg[2];
  F4 breg[2][2];
  const int NK = HID / 32;

  // prologue
#pragma unroll
  for (int i = 0; i < 2; ++i) { areg[i] = gload_u4(aptr[i]); aptr[i] += 32; }
#pragma unroll
  for (int i = 0; i < 2; ++i) {
    breg[i][0].v = gload_f4(bptr[i]);
    breg[i][1].v = gload_f4(bptr[i] + DIMD);
    bptr[i] += 32 * DIMD;
  }
#pragma unroll
  for (int i = 0; i < 2; ++i) *(u32x4*)&As[0][aoff[i]] = areg[i];
#pragma unroll
  for (int i = 0; i < 2; ++i)
#pragma unroll
    for (int j = 0; j < 4; ++j)
      *(uint32_t*)&Bs[0][boff[i][j]] = pack2bf(breg[i][0].a[j], breg[i][1].a[j]);
  __syncthreads();

  for (int kt = 0; kt < NK; ++kt) {
    int cur = kt & 1;
    bool hasNext = (kt + 1) < NK;
    if (hasNext) {
#pragma unroll
      for (int i = 0; i < 2; ++i) { areg[i] = gload_u4(aptr[i]); aptr[i] += 32; }
#pragma unroll
      for (int i = 0; i < 2; ++i) {
        breg[i][0].v = gload_f4(bptr[i]);
        breg[i][1].v = gload_f4(bptr[i] + DIMD);
        bptr[i] += 32 * DIMD;
      }
    }
    v16bf fb[2];
#pragma unroll
    for (int ns = 0; ns < 2; ++ns)
      fb[ns] = load_frag16(Bs[cur], waveN + ns * 16 + laneM, kOff);
#pragma unroll
    for (int ms = 0; ms < 4; ++ms) {
      v16bf fa = load_frag16(As[cur], waveM + ms * 16 + laneM, kOff);
#pragma unroll
      for (int ns = 0; ns < 2; ++ns)
        acc[ms][ns] = __builtin_amdgcn_wmma_f32_16x16x32_bf16(
            false, fa, false, fb[ns], (short)0, acc[ms][ns], false, false);
    }
    if (hasNext) {
      int nxt = cur ^ 1;
#pragma unroll
      for (int i = 0; i < 2; ++i) *(u32x4*)&As[nxt][aoff[i]] = areg[i];
#pragma unroll
      for (int i = 0; i < 2; ++i)
#pragma unroll
        for (int j = 0; j < 4; ++j)
          *(uint32_t*)&Bs[nxt][boff[i][j]] = pack2bf(breg[i][0].a[j], breg[i][1].a[j]);
    }
    __syncthreads();
  }

  // epilogue: + b2, scale by routing weight, atomic scatter-add to out
#pragma unroll
  for (int ms = 0; ms < 4; ++ms) {
#pragma unroll
    for (int ns = 0; ns < 2; ++ns) {
      int nG = n0 + waveN + ns * 16 + laneM;
      float bb = b2[e * DIMD + nG];
      int rBase = waveM + ms * 16 + laneHalf * 8;
#pragma unroll
      for (int r = 0; r < 8; ++r) {
        int row = rBase + r;
        if (m0 + row < cnt) {
          float val = (acc[ms][ns][r] + bb) * sw[row];
          atomicAdd(&out[(size_t)stok[row] * DIMD + nG], val);
        }
      }
    }
  }
}

// ---------------- launcher ----------------
extern "C" void kernel_launch(void* const* d_in, const int* in_sizes, int n_in,
                              void* d_out, int out_size, void* d_ws, size_t ws_size,
                              hipStream_t stream) {
  const float* x    = (const float*)d_in[0];
  const float* Wg   = (const float*)d_in[1];
  const float* W1   = (const float*)d_in[2];
  const float* b1   = (const float*)d_in[3];
  const float* ln_g = (const float*)d_in[4];
  const float* ln_b = (const float*)d_in[5];
  const float* W2   = (const float*)d_in[6];
  const float* b2   = (const float*)d_in[7];
  float* out = (float*)d_out;

  char* ws = (char*)d_ws;
  int*   counts = (int*)ws;                      // 16
  int*   cursor = counts + 16;                   // 16
  int*   seg    = cursor + 16;                   // 16
  int*   sel    = (int*)  (ws + 1024);           // 4096
  float* wts    = (float*)(ws + 1024 + 16384);   // 4096
  int*   ptok   = (int*)  (ws + 1024 + 2 * 16384);
  float* pw     = (float*)(ws + 1024 + 3 * 16384);
  int*   pexp   = (int*)  (ws + 1024 + 4 * 16384);
  float*  hbuf  = (float*)(ws + (1 << 17));                         // 32 MB
  __bf16* hb    = (__bf16*)(ws + (1 << 17) + (size_t)(32u << 20));  // 16 MB

  moe_init_kernel<<<(T_TOK * DIMD + 255) / 256, 256, 0, stream>>>(out, counts, cursor);
  moe_route_kernel<<<T_TOK / 16, 256, 0, stream>>>(x, Wg, sel, wts, counts);
  moe_prefix_kernel<<<1, 32, 0, stream>>>(counts, seg);
  moe_scatter_kernel<<<(T_TOK * 2) / 256, 256, 0, stream>>>(sel, wts, seg, cursor,
                                                            ptok, pw, pexp);
  moe_gemm1_kernel<<<dim3(HID / 128, 16, NE), 256, 0, stream>>>(
      x, W1, b1, counts, seg, ptok, hbuf);
  moe_ln_kernel<<<T_TOK * 2, 256, 0, stream>>>(hbuf, ln_g, ln_b, pexp, hb);
  moe_gemm2_kernel<<<dim3(DIMD / 128, 16, NE), 256, 0, stream>>>(
      hb, W2, b2, counts, seg, ptok, pw, out);
}
